// Block_ViT_63367947485681
// MI455X (gfx1250) — matrix operand order repaired and checked by reading, not compiled, not run on hardware
//
#include <hip/hip_runtime.h>
#include <math.h>

// ---------------------------------------------------------------------------
// Problem constants (from the reference)
// ---------------------------------------------------------------------------
#define B_    16
#define N_    196
#define E_    512
#define H_    8
#define HD_   64
#define C_    2048
#define HDC_  256
#define N4_   784
#define F_    2048   // 4*E

// ---------------------------------------------------------------------------
// WMMA types (CDNA5 / gfx1250, wave32)
// ---------------------------------------------------------------------------
typedef __attribute__((ext_vector_type(16))) __bf16 v16bf;
typedef __attribute__((ext_vector_type(8)))  float  v8f;

union Frag16 { v16bf v; uint4 q[2]; };
union BF2    { unsigned u; __bf16 h[2]; };

__device__ __forceinline__ unsigned pk_bf2(float lo, float hi) {
  BF2 t; t.h[0] = (__bf16)lo; t.h[1] = (__bf16)hi; return t.u;
}

// clamped, unconditional float4 load; value zeroed by cndmask when OOB
__device__ __forceinline__ float4 ld4z(const float* base, long idx, bool valid) {
  float4 v = *(const float4*)(base + idx);
  v.x = valid ? v.x : 0.0f; v.y = valid ? v.y : 0.0f;
  v.z = valid ? v.z : 0.0f; v.w = valid ? v.w : 0.0f;
  return v;
}

// ---------------------------------------------------------------------------
// Generic batched WMMA GEMM:  D = epilogue( A[M,K] * B[K,N] )
//   TA: A stored [K,M] (read transposed)    TB: B stored [N,K]
//   EPI: 0=none  1=bias+exact GeLU  2=bias+residual  3=residual
//   batch z decomposed: i2 = z % div2 ; t = z / div2 ; i1 = t % div1 ; i0 = t/div1
//   Requirements (hold for every call in this network): M,N,K multiples of 4,
//   all base offsets / leading dims multiples of 4 floats.
// ---------------------------------------------------------------------------
struct GemmParams {
  const float* A; const float* Bm; float* D;
  const float* bias; const float* R;
  int M, N, K;
  int lda, ldb, ldd;
  int div1, div2;
  long sA0, sA1, sA2;
  long sB0, sB1, sB2;
  long sD0, sD1, sD2;
  long sb0, sb1, sb2;
};

#define BM 64
#define BN 64
#define BK 64
#define LPAD 8    // LDS row pad (elements): row stride 144B, keeps 16B alignment

template<bool TA, bool TB, int EPI>
__global__ __launch_bounds__(128)
void gemm_wmma(GemmParams p) {
  __shared__ __bf16 As[BM][BK + LPAD];   // As[m][k]
  __shared__ __bf16 Bt[BN][BK + LPAD];   // B tile TRANSPOSED in LDS: Bt[n][k]

  const int z  = blockIdx.z;
  const int i2 = z % p.div2;
  const int t1 = z / p.div2;
  const int i1 = t1 % p.div1;
  const int i0 = t1 / p.div1;

  const float* Ab = p.A  + (long)i0 * p.sA0 + (long)i1 * p.sA1 + (long)i2 * p.sA2;
  const float* Bb = p.Bm + (long)i0 * p.sB0 + (long)i1 * p.sB1 + (long)i2 * p.sB2;
  const long   dOff = (long)i0 * p.sD0 + (long)i1 * p.sD1 + (long)i2 * p.sD2;
  float* Db = p.D + dOff;
  const float* biasb = (EPI == 1 || EPI == 2)
      ? (p.bias + (long)i0 * p.sb0 + (long)i1 * p.sb1 + (long)i2 * p.sb2) : nullptr;
  const float* Rb = (EPI == 2 || EPI == 3) ? (p.R + dOff) : nullptr;

  const int m0   = blockIdx.y * BM;
  const int n0   = blockIdx.x * BN;
  const int tid  = threadIdx.x;
  const int wave = tid >> 5;
  const int lane = tid & 31;
  const int wm   = (wave >> 1) * 32;
  const int wn   = (wave & 1) * 32;
  const int lh   = lane >> 4;     // lane half (0/1)
  const int l15  = lane & 15;
  const int kb   = lh * 8;        // A frag: lane-half K offset
  const int kbb  = lh * 16;       // unused; B frag uses kb, same as A

  v8f acc[2][2];
  acc[0][0] = (v8f){0,0,0,0,0,0,0,0}; acc[0][1] = (v8f){0,0,0,0,0,0,0,0};
  acc[1][0] = (v8f){0,0,0,0,0,0,0,0}; acc[1][1] = (v8f){0,0,0,0,0,0,0,0};

  for (int k0 = 0; k0 < p.K; k0 += BK) {
    // ---- gather A & B tiles into registers (unconditional clamped float4) ----
    float4 av[8], bv[8];
    #pragma unroll
    for (int j = 0; j < 8; ++j) {
      const int q = tid + 128 * j;                 // 1024 float4s per tile
      if (!TA) {            // A[m][k], contiguous k
        const int r = q >> 4, kkb = (q & 15) * 4;
        const int gm = m0 + r, gk = k0 + kkb;
        const bool val = (gm < p.M) && (gk < p.K);
        const long idx = (long)(gm < p.M ? gm : 0) * p.lda + (gk < p.K ? gk : 0);
        av[j] = ld4z(Ab, idx, val);
      } else {              // A[k][m], contiguous m
        const int kk = q >> 4, rb = (q & 15) * 4;
        const int gk = k0 + kk, gm = m0 + rb;
        const bool val = (gm < p.M) && (gk < p.K);
        const long idx = (long)(gk < p.K ? gk : 0) * p.lda + (gm < p.M ? gm : 0);
        av[j] = ld4z(Ab, idx, val);
      }
    }
    #pragma unroll
    for (int j = 0; j < 8; ++j) {
      const int q = tid + 128 * j;
      if (!TB) {            // B[k][n], contiguous n
        const int kk = q >> 4, cb = (q & 15) * 4;
        const int gk = k0 + kk, gn = n0 + cb;
        const bool val = (gk < p.K) && (gn < p.N);
        const long idx = (long)(gk < p.K ? gk : 0) * p.ldb + (gn < p.N ? gn : 0);
        bv[j] = ld4z(Bb, idx, val);
      } else {              // B[n][k], contiguous k
        const int c = q >> 4, kkb2 = (q & 15) * 4;
        const int gn = n0 + c, gk = k0 + kkb2;
        const bool val = (gn < p.N) && (gk < p.K);
        const long idx = (long)(gn < p.N ? gn : 0) * p.ldb + (gk < p.K ? gk : 0);
        bv[j] = ld4z(Bb, idx, val);
      }
    }
    __syncthreads();   // previous iteration's LDS reads complete

    // ---- store tiles to LDS as bf16 ----
    #pragma unroll
    for (int j = 0; j < 8; ++j) {
      const int q = tid + 128 * j;
      if (!TA) {
        const int r = q >> 4, kkb = (q & 15) * 4;
        unsigned* dst = (unsigned*)&As[r][kkb];              // 8B-aligned packed store
        dst[0] = pk_bf2(av[j].x, av[j].y);
        dst[1] = pk_bf2(av[j].z, av[j].w);
      } else {
        const int kk = q >> 4, rb = (q & 15) * 4;
        As[rb + 0][kk] = (__bf16)av[j].x;
        As[rb + 1][kk] = (__bf16)av[j].y;
        As[rb + 2][kk] = (__bf16)av[j].z;
        As[rb + 3][kk] = (__bf16)av[j].w;
      }
    }
    #pragma unroll
    for (int j = 0; j < 8; ++j) {
      const int q = tid + 128 * j;
      if (!TB) {
        const int kk = q >> 4, cb = (q & 15) * 4;
        Bt[cb + 0][kk] = (__bf16)bv[j].x;
        Bt[cb + 1][kk] = (__bf16)bv[j].y;
        Bt[cb + 2][kk] = (__bf16)bv[j].z;
        Bt[cb + 3][kk] = (__bf16)bv[j].w;
      } else {
        const int c = q >> 4, kkb2 = (q & 15) * 4;
        unsigned* dst = (unsigned*)&Bt[c][kkb2];
        dst[0] = pk_bf2(bv[j].x, bv[j].y);
        dst[1] = pk_bf2(bv[j].z, bv[j].w);
      }
    }
    __syncthreads();

    // ---- 2 k-substeps x (2x2) WMMA; all fragment loads are ds_load_b128 ----
    #pragma unroll
    for (int ks = 0; ks < BK; ks += 32) {
      Frag16 a0, a1, b0, b1;
      const int r0 = wm + l15, r1 = wm + 16 + l15;
      a0.q[0] = *(const uint4*)&As[r0][ks + kb];
      a0.q[1] = *(const uint4*)&As[r0][ks + 16 + kb];
      a1.q[0] = *(const uint4*)&As[r1][ks + kb];
      a1.q[1] = *(const uint4*)&As[r1][ks + 16 + kb];
      const int c0 = wn + l15, c1 = wn + 16 + l15;
      b0.q[0] = *(const uint4*)&Bt[c0][ks + kb];
      b0.q[1] = *(const uint4*)&Bt[c0][ks + 16 + kb];
      b1.q[0] = *(const uint4*)&Bt[c1][ks + kb];
      b1.q[1] = *(const uint4*)&Bt[c1][ks + 16 + kb];
      acc[0][0] = __builtin_amdgcn_wmma_f32_16x16x32_bf16(false, a0.v, false, b0.v, (short)0, acc[0][0], false, false);
      acc[0][1] = __builtin_amdgcn_wmma_f32_16x16x32_bf16(false, a0.v, false, b1.v, (short)0, acc[0][1], false, false);
      acc[1][0] = __builtin_amdgcn_wmma_f32_16x16x32_bf16(false, a1.v, false, b0.v, (short)0, acc[1][0], false, false);
      acc[1][1] = __builtin_amdgcn_wmma_f32_16x16x32_bf16(false, a1.v, false, b1.v, (short)0, acc[1][1], false, false);
    }
    __syncthreads();
  }

  // ---- epilogue + store (C/D layout: lane-half -> rows 0-7 / 8-15) ----
  #pragma unroll
  for (int tm = 0; tm < 2; ++tm) {
    #pragma unroll
    for (int tn = 0; tn < 2; ++tn) {
      const int col = n0 + wn + tn * 16 + l15;
      const int rowbase = m0 + wm + tm * 16 + lh * 8;
      if (col < p.N) {
        #pragma unroll
        for (int r = 0; r < 8; ++r) {
          const int m = rowbase + r;
          if (m < p.M) {
            float v = acc[tm][tn][r];
            if (EPI == 1) { v += biasb[col]; v = 0.5f * v * (1.0f + erff(v * 0.70710678118654752f)); }
            else if (EPI == 2) { v += biasb[col] + Rb[(long)m * p.ldd + col]; }
            else if (EPI == 3) { v += Rb[(long)m * p.ldd + col]; }
            Db[(long)m * p.ldd + col] = v;
          }
        }
      }
    }
  }
}

// ---------------------------------------------------------------------------
// Block reductions (256-thread blocks)
// ---------------------------------------------------------------------------
__device__ __forceinline__ float block_sum(float v, float* red) {
  int tid = threadIdx.x;
  red[tid] = v; __syncthreads();
  for (int s = 128; s > 0; s >>= 1) { if (tid < s) red[tid] += red[tid + s]; __syncthreads(); }
  float r = red[0]; __syncthreads();
  return r;
}
__device__ __forceinline__ float block_max(float v, float* red) {
  int tid = threadIdx.x;
  red[tid] = v; __syncthreads();
  for (int s = 128; s > 0; s >>= 1) { if (tid < s) red[tid] = fmaxf(red[tid], red[tid + s]); __syncthreads(); }
  float r = red[0]; __syncthreads();
  return r;
}

// ---------------------------------------------------------------------------
// LN over the 4 branch embeddings (writes cx) + raw copy into hbuf (residual)
// ---------------------------------------------------------------------------
__global__ __launch_bounds__(256)
void ln4_copy_kernel(const float* __restrict__ e1, const float* __restrict__ e2,
                     const float* __restrict__ e3, const float* __restrict__ e4,
                     const float* __restrict__ g, const float* __restrict__ bt,
                     float* __restrict__ cx, float* __restrict__ hb) {
  __shared__ float red[256];
  const int row = blockIdx.x;                  // [0, 4*B*N)
  const int gi  = row / (B_ * N_);
  const int rn  = row % (B_ * N_);
  const float* src = (gi == 0) ? e1 : (gi == 1) ? e2 : (gi == 2) ? e3 : e4;
  src += (long)rn * E_;
  const int tid = threadIdx.x;
  const float x0 = src[tid], x1 = src[tid + 256];
  const float mean = block_sum(x0 + x1, red) * (1.0f / E_);
  const float d0 = x0 - mean, d1 = x1 - mean;
  const float rstd = rsqrtf(block_sum(d0 * d0 + d1 * d1, red) * (1.0f / E_) + 1e-6f);
  const long o = (long)row * E_;
  cx[o + tid]       = d0 * rstd * g[gi * E_ + tid]       + bt[gi * E_ + tid];
  cx[o + tid + 256] = d1 * rstd * g[gi * E_ + tid + 256] + bt[gi * E_ + tid + 256];
  hb[o + tid] = x0; hb[o + tid + 256] = x1;
}

// LN over concat(emb1..4) along channels -> ec [B,N,C]
__global__ __launch_bounds__(256)
void lnC_kernel(const float* __restrict__ e1, const float* __restrict__ e2,
                const float* __restrict__ e3, const float* __restrict__ e4,
                const float* __restrict__ g, const float* __restrict__ bt,
                float* __restrict__ ec) {
  __shared__ float red[256];
  const int row = blockIdx.x;                  // [0, B*N)
  const int tid = threadIdx.x;
  float x[8]; float s = 0.f;
  #pragma unroll
  for (int j = 0; j < 8; ++j) {
    const int c = tid + 256 * j;
    const int gg = c >> 9, off = c & 511;
    const float* sp = (gg == 0) ? e1 : (gg == 1) ? e2 : (gg == 2) ? e3 : e4;
    x[j] = sp[(long)row * E_ + off];
    s += x[j];
  }
  const float mean = block_sum(s, red) * (1.0f / C_);
  float vs = 0.f;
  #pragma unroll
  for (int j = 0; j < 8; ++j) { const float d = x[j] - mean; vs += d * d; }
  const float rstd = rsqrtf(block_sum(vs, red) * (1.0f / C_) + 1e-6f);
  #pragma unroll
  for (int j = 0; j < 8; ++j) {
    const int c = tid + 256 * j;
    ec[(long)row * C_ + c] = (x[j] - mean) * rstd * g[c] + bt[c];
  }
}

// LN over contiguous [4,B,N,E] rows (FFN pre-norm)
__global__ __launch_bounds__(256)
void ln_rows_kernel(const float* __restrict__ xin, const float* __restrict__ g,
                    const float* __restrict__ bt, float* __restrict__ y) {
  __shared__ float red[256];
  const int row = blockIdx.x;                  // [0, 4*B*N)
  const int gi  = row / (B_ * N_);
  const float* src = xin + (long)row * E_;
  const int tid = threadIdx.x;
  const float x0 = src[tid], x1 = src[tid + 256];
  const float mean = block_sum(x0 + x1, red) * (1.0f / E_);
  const float d0 = x0 - mean, d1 = x1 - mean;
  const float rstd = rsqrtf(block_sum(d0 * d0 + d1 * d1, red) * (1.0f / E_) + 1e-6f);
  const long o = (long)row * E_;
  y[o + tid]       = d0 * rstd * g[gi * E_ + tid]       + bt[gi * E_ + tid];
  y[o + tid + 256] = d1 * rstd * g[gi * E_ + tid + 256] + bt[gi * E_ + tid + 256];
}

// scaled row softmax over 196-wide rows (channel attention)
__global__ __launch_bounds__(256)
void softmax196_kernel(float* __restrict__ S) {
  __shared__ float red[256];
  const long row = blockIdx.x;                 // [0, B*H*N)
  float* p = S + row * N_;
  const int tid = threadIdx.x;
  const float v = (tid < N_) ? p[tid] * 0.0625f : -3.4e38f;   // 1/sqrt(256)
  const float mx = block_max(v, red);
  const float e  = (tid < N_) ? __expf(v - mx) : 0.0f;
  const float sm = block_sum(e, red);
  if (tid < N_) p[tid] = e / sm;
}

// KV_S[b, g*N+n, e] = That[b, n, g*E+e]
__global__ __launch_bounds__(256)
void kvs_permute_kernel(const float* __restrict__ That, float* __restrict__ KVS) {
  const long idx = (long)blockIdx.x * 256 + threadIdx.x;
  const long total = (long)B_ * N4_ * E_;
  if (idx >= total) return;
  const int e = (int)(idx % E_); const long t = idx / E_;
  const int m = (int)(t % N4_);  const int b = (int)(t / N4_);
  const int g = m / N_, n = m % N_;
  KVS[idx] = That[((long)b * N_ + n) * C_ + g * E_ + e];
}

// InstanceNorm stats per (b,h): mean + rstd over N x 4N score map
__global__ __launch_bounds__(256)
void inorm_stats_kernel(const float* __restrict__ attn, float* __restrict__ st) {
  __shared__ float red[256];
  const int bh = blockIdx.x;                   // [0, B*H)
  const float* p = attn + (long)bh * (N_ * N4_);
  float s = 0.f, ss = 0.f;
  for (int i = threadIdx.x; i < N_ * N4_; i += 256) { const float v = p[i]; s += v; ss += v * v; }
  const float S1 = block_sum(s, red);
  const float S2 = block_sum(ss, red);
  if (threadIdx.x == 0) {
    const float inv = 1.0f / (float)(N_ * N4_);
    const float mean = S1 * inv;
    const float var  = S2 * inv - mean * mean;
    st[bh * 2] = mean;
    st[bh * 2 + 1] = rsqrtf(var + 1e-5f);
  }
}

// normalized row softmax over 784-wide rows (spatial attention)
__global__ __launch_bounds__(256)
void inorm_softmax_kernel(float* __restrict__ attn, const float* __restrict__ st) {
  __shared__ float red[256];
  const int row = blockIdx.x;                  // [0, B*H*N)
  const int bh  = row / N_;
  const float mean = st[bh * 2], rstd = st[bh * 2 + 1];
  float* p = attn + (long)row * N4_;
  const int tid = threadIdx.x;
  float v[4]; float mx = -3.4e38f;
  #pragma unroll
  for (int j = 0; j < 4; ++j) {
    const int c = tid + 256 * j;
    v[j] = (c < N4_) ? (p[c] - mean) * rstd : -3.4e38f;
    mx = fmaxf(mx, v[j]);
  }
  mx = block_max(mx, red);
  float e[4]; float s = 0.f;
  #pragma unroll
  for (int j = 0; j < 4; ++j) {
    const int c = tid + 256 * j;
    e[j] = (c < N4_) ? __expf(v[j] - mx) : 0.0f;
    s += e[j];
  }
  s = block_sum(s, red);
  #pragma unroll
  for (int j = 0; j < 4; ++j) {
    const int c = tid + 256 * j;
    if (c < N4_) p[c] = e[j] / s;
  }
}

// merge ctx [4,B,H,N,HD] -> cmerged [4,B,N,E]
// branch 0: head-interleave permute; branches 1-3: head-major direct flatten
__global__ __launch_bounds__(256)
void merge_kernel(const float* __restrict__ ctx, float* __restrict__ cmg) {
  const long idx = (long)blockIdx.x * 256 + threadIdx.x;
  const long total = 4L * B_ * N_ * E_;
  if (idx >= total) return;
  const int e = (int)(idx % E_); long t = idx / E_;
  const int n = (int)(t % N_);   t /= N_;
  const int b = (int)(t % B_);
  const int g = (int)(t / B_);
  float val;
  if (g == 0) {
    const int h = e / HD_, d = e % HD_;
    val = ctx[((((long)b) * H_ + h) * N_ + n) * HD_ + d];
  } else {
    const long f = (long)n * E_ + e;             // flat over [H,N,HD]
    const int h  = (int)(f / (N_ * HD_));
    const long r = f % (N_ * HD_);
    const int n2 = (int)(r / HD_), d = (int)(r % HD_);
    val = ctx[((((long)g * B_ + b) * H_ + h) * N_ + n2) * HD_ + d];
  }
  cmg[idx] = val;
}

// ---------------------------------------------------------------------------
// Host-side launch helpers
// ---------------------------------------------------------------------------
static GemmParams mkp(const float* A, const float* Bm, float* D,
                      const float* bias, const float* R,
                      int M, int N, int K, int lda, int ldb, int ldd,
                      int div1, int div2,
                      long sA0, long sA1, long sA2,
                      long sB0, long sB1, long sB2,
                      long sD0, long sD1, long sD2,
                      long sb0 = 0, long sb1 = 0, long sb2 = 0) {
  GemmParams p;
  p.A = A; p.Bm = Bm; p.D = D; p.bias = bias; p.R = R;
  p.M = M; p.N = N; p.K = K; p.lda = lda; p.ldb = ldb; p.ldd = ldd;
  p.div1 = div1; p.div2 = div2;
  p.sA0 = sA0; p.sA1 = sA1; p.sA2 = sA2;
  p.sB0 = sB0; p.sB1 = sB1; p.sB2 = sB2;
  p.sD0 = sD0; p.sD1 = sD1; p.sD2 = sD2;
  p.sb0 = sb0; p.sb1 = sb1; p.sb2 = sb2;
  return p;
}

template<bool TA, bool TB, int EPI>
static void launch_gemm(const GemmParams& p, int batches, hipStream_t s) {
  dim3 grid((p.N + BN - 1) / BN, (p.M + BM - 1) / BM, batches);
  gemm_wmma<TA, TB, EPI><<<grid, dim3(128), 0, s>>>(p);
}

extern "C" void kernel_launch(void* const* d_in, const int* in_sizes, int n_in,
                              void* d_out, int out_size, void* d_ws, size_t ws_size,
                              hipStream_t stream) {
  (void)in_sizes; (void)n_in; (void)out_size;

  const float* emb1    = (const float*)d_in[0];
  const float* emb2    = (const float*)d_in[1];
  const float* emb3    = (const float*)d_in[2];
  const float* emb4    = (const float*)d_in[3];
  const float* lnA_g   = (const float*)d_in[4];
  const float* lnA_b   = (const float*)d_in[5];
  const float* lnC_g   = (const float*)d_in[6];
  const float* lnC_b   = (const float*)d_in[7];
  const float* Wq_c    = (const float*)d_in[8];
  const float* Wk_c    = (const float*)d_in[9];
  const float* Wv_c    = (const float*)d_in[10];
  const float* Wo_c    = (const float*)d_in[11];
  const float* q_w     = (const float*)d_in[12];
  const float* k_w     = (const float*)d_in[13];
  const float* v_w     = (const float*)d_in[14];
  const float* out_w   = (const float*)d_in[15];
  const float* lnF_g   = (const float*)d_in[16];
  const float* lnF_b   = (const float*)d_in[17];
  const float* fc1_w   = (const float*)d_in[18];
  const float* fc1_b   = (const float*)d_in[19];
  const float* fc2_w   = (const float*)d_in[20];
  const float* fc2_b   = (const float*)d_in[21];
  float* outp = (float*)d_out;                 // [4,B,N,E] concatenated

  // ---- workspace layout (floats) ----
  const long SZ  = 4L * B_ * N_ * E_;          // 6,422,528  (== B*N*C)
  const long SSC = (long)B_ * H_ * N_ * N_;    // 4,917,248
  const long SAT = (long)B_ * H_ * N_ * N4_;   // 19,668,992 (per-g score map)
  const long SFF = 4L * B_ * N_ * F_;          // 25,690,112
  float* ws = (float*)d_ws;
  long o = 0;
  float* cx    = ws + o; o += SZ;
  float* ec    = ws + o; o += SZ;
  float* qc    = ws + o; o += SZ;
  float* kc    = ws + o; o += SZ;
  float* vc    = ws + o; o += SZ;
  float* Sc    = ws + o; o += SSC;
  float* Tpre  = ws + o; o += SZ;
  float* That  = ws + o; o += SZ;
  float* KVS   = ws + o; o += SZ;
  float* Qt    = ws + o; o += SZ;
  float* Kt    = ws + o; o += SZ;
  float* Vt    = ws + o; o += SZ;
  float* ctxb  = ws + o; o += SZ;
  float* cmg   = ws + o; o += SZ;
  float* hbuf  = ws + o; o += SZ;
  float* xbuf  = ws + o; o += SZ;
  float* ffb   = ws + o; o += SFF;
  float* attng = ws + o; o += SAT;
  float* ist   = ws + o; o += 2L * B_ * H_;
  if (ws_size < (size_t)o * sizeof(float)) return;   // scratch too small: bail deterministically

  const int BNrows = B_ * N_;                  // 3136
  const long sBNE  = (long)B_ * N_ * E_;       // 1,605,632 (per-g stride of [4,B,N,E])
  const long sNE   = (long)N_ * E_;            // 100,352
  const long sNC   = (long)N_ * C_;            // 401,408

  // 1) per-branch LN -> cx ; copy embs -> hbuf (residual base)
  ln4_copy_kernel<<<4 * BNrows, 256, 0, stream>>>(emb1, emb2, emb3, emb4, lnA_g, lnA_b, cx, hbuf);
  // 2) concat LN -> ec [B,N,C]
  lnC_kernel<<<BNrows, 256, 0, stream>>>(emb1, emb2, emb3, emb4, lnC_g, lnC_b, ec);

  // 3) channel q/k/v : [3136,2048] @ [2048,2048]
  launch_gemm<false,false,0>(mkp(ec, Wq_c, qc, nullptr, nullptr, BNrows, C_, C_, C_, C_, C_,
                                 1,1, 0,0,0, 0,0,0, 0,0,0), 1, stream);
  launch_gemm<false,false,0>(mkp(ec, Wk_c, kc, nullptr, nullptr, BNrows, C_, C_, C_, C_, C_,
                                 1,1, 0,0,0, 0,0,0, 0,0,0), 1, stream);
  launch_gemm<false,false,0>(mkp(ec, Wv_c, vc, nullptr, nullptr, BNrows, C_, C_, C_, C_, C_,
                                 1,1, 0,0,0, 0,0,0, 0,0,0), 1, stream);

  // 4) channel scores: per (b,h)  q[196,256] @ k^T  -> Sc [B,H,196,196]
  launch_gemm<false,true,0>(mkp(qc, kc, Sc, nullptr, nullptr, N_, N_, HDC_, C_, C_, N_,
                                B_, H_,
                                0, sNC, HDC_,
                                0, sNC, HDC_,
                                0, (long)H_*N_*N_, (long)N_*N_), B_*H_, stream);
  // 5) scale + softmax
  softmax196_kernel<<<B_ * H_ * N_, 256, 0, stream>>>(Sc);
  // 6) ctx: per (b,h)  S[196,196] @ v[196,256] -> Tpre [B,N,C]
  launch_gemm<false,false,0>(mkp(Sc, vc, Tpre, nullptr, nullptr, N_, HDC_, N_, N_, C_, C_,
                                 B_, H_,
                                 0, (long)H_*N_*N_, (long)N_*N_,
                                 0, sNC, HDC_,
                                 0, sNC, HDC_), B_*H_, stream);
  // 7) output projection of channel attention
  launch_gemm<false,false,0>(mkp(Tpre, Wo_c, That, nullptr, nullptr, BNrows, C_, C_, C_, C_, C_,
                                 1,1, 0,0,0, 0,0,0, 0,0,0), 1, stream);
  // 8) reshape to KV_S [B,4N,E]
  kvs_permute_kernel<<<(int)(((long)B_*N4_*E_ + 255) / 256), 256, 0, stream>>>(That, KVS);

  // 9) token-mix Q: per (g,b)  q_w[g]^T[196,196] @ cx[g,b][196,512]
  launch_gemm<true,false,0>(mkp(q_w, cx, Qt, nullptr, nullptr, N_, E_, N_, N_, E_, E_,
                                4, B_,
                                0, (long)N_*N_, 0,
                                0, sBNE, sNE,
                                0, sBNE, sNE), 4*B_, stream);
  // 10) token-mix K/V: per b  k_w^T[784,784] @ KVS[b][784,512]
  launch_gemm<true,false,0>(mkp(k_w, KVS, Kt, nullptr, nullptr, N4_, E_, N4_, N4_, E_, E_,
                                1, B_,
                                0, 0, 0,
                                0, 0, (long)N4_*E_,
                                0, 0, (long)N4_*E_), B_, stream);
  launch_gemm<true,false,0>(mkp(v_w, KVS, Vt, nullptr, nullptr, N4_, E_, N4_, N4_, E_, E_,
                                1, B_,
                                0, 0, 0,
                                0, 0, (long)N4_*E_,
                                0, 0, (long)N4_*E_), B_, stream);

  // 11) spatial attention, chunked per branch g to bound scratch
  for (int g = 0; g < 4; ++g) {
    // scores: per (b,h)  Qh[196,64] @ Kh^T -> attng [B,H,196,784]
    launch_gemm<false,true,0>(mkp(Qt + (long)g * sBNE, Kt, attng, nullptr, nullptr,
                                  N_, N4_, HD_, E_, E_, N4_,
                                  B_, H_,
                                  0, sNE, HD_,
                                  0, (long)N4_*E_, HD_,
                                  0, (long)H_*N_*N4_, (long)N_*N4_), B_*H_, stream);
    inorm_stats_kernel<<<B_ * H_, 256, 0, stream>>>(attng, ist);
    inorm_softmax_kernel<<<B_ * H_ * N_, 256, 0, stream>>>(attng, ist);
    // ctx: per (b,h)  p[196,784] @ Vh[784,64] -> ctxb [4,B,H,N,HD]
    launch_gemm<false,false,0>(mkp(attng, Vt, ctxb + (long)g * B_ * H_ * N_ * HD_,
                                   nullptr, nullptr,
                                   N_, HD_, N4_, N4_, E_, HD_,
                                   B_, H_,
                                   0, (long)H_*N_*N4_, (long)N_*N4_,
                                   0, (long)N4_*E_, HD_,
                                   0, (long)H_*N_*HD_, (long)N_*HD_), B_*H_, stream);
  }

  // 12) merge ctx -> cmerged [4,B,N,E]
  merge_kernel<<<(int)((4L*B_*N_*E_ + 255) / 256), 256, 0, stream>>>(ctxb, cmg);

  // 13) out projection + residual 1:  hbuf = embs + cmg @ out_w[g]
  launch_gemm<false,false,3>(mkp(cmg, out_w, hbuf, nullptr, hbuf, BNrows, E_, E_, E_, E_, E_,
                                 4, 1,
                                 0, sBNE, 0,
                                 0, (long)E_*E_, 0,
                                 0, sBNE, 0), 4, stream);

  // 14) FFN pre-norm
  ln_rows_kernel<<<4 * BNrows, 256, 0, stream>>>(hbuf, lnF_g, lnF_b, xbuf);

  // 15) fc1 + bias + exact GeLU
  launch_gemm<false,false,1>(mkp(xbuf, fc1_w, ffb, fc1_b, nullptr, BNrows, F_, E_, E_, F_, F_,
                                 4, 1,
                                 0, sBNE, 0,
                                 0, (long)E_*F_, 0,
                                 0, (long)BNrows*F_, 0,
                                 0, F_, 0), 4, stream);

  // 16) fc2 + bias + residual 2 -> d_out [4,B,N,E]
  launch_gemm<false,false,2>(mkp(ffb, fc2_w, outp, fc2_b, hbuf, BNrows, E_, F_, F_, E_, E_,
                                 4, 1,
                                 0, (long)BNrows*F_, 0,
                                 0, (long)F_*E_, 0,
                                 0, sBNE, 0,
                                 0, E_, 0), 4, stream);
}
